// RNN_42219528520009
// MI455X (gfx1250) — compile-verified, matrix operand order checked by reading, not run on hardware
//
#include <hip/hip_runtime.h>
#include <hip/hip_bf16.h>
#include <stdint.h>

#define Bsz 64
#define Tsz 512
#define Hsz 1024
#define Isz 256
#define Osz 128

typedef __attribute__((ext_vector_type(16))) __bf16 bf16x16;
typedef __attribute__((ext_vector_type(8)))  __bf16 bf16x8;
typedef __attribute__((ext_vector_type(8)))  float  f32x8;
typedef __attribute__((ext_vector_type(4)))  unsigned int u32x4;
typedef __attribute__((ext_vector_type(8)))  int i32x8;
typedef __attribute__((ext_vector_type(4)))  int i32x4;

#if defined(__HIP_DEVICE_COMPILE__) && \
    __has_builtin(__builtin_amdgcn_tensor_load_to_lds) && \
    __has_builtin(__builtin_amdgcn_s_wait_tensorcnt)
#define USE_TDM 1
#endif

// ---- fragment loaders (ISA 16-bit A/B layout: row = lane&15, K chunks
// [8h, 8h+8) and [16+8h, 16+8h+8) per half-wave h = lane>>4) ----
__device__ __forceinline__ bf16x16 frag_ld_bf16(const __bf16* __restrict__ p, int ld,
                                                int rbase, int kbase, int lane) {
  const int r  = rbase + (lane & 15);
  const int k0 = kbase + ((lane >> 4) << 3);
  const __bf16* q = p + (size_t)r * ld + k0;
  bf16x16 f;
  *(bf16x8*)&f       = *(const bf16x8*)q;
  *((bf16x8*)&f + 1) = *(const bf16x8*)(q + 16);
  return f;
}

__device__ __forceinline__ bf16x16 frag_ld_f32(const float* __restrict__ p, int ld,
                                               int rbase, int kbase, int lane) {
  const int r  = rbase + (lane & 15);
  const int k0 = kbase + ((lane >> 4) << 3);
  const float* q = p + (size_t)r * ld + k0;
  bf16x16 f;
#pragma unroll
  for (int e = 0; e < 8; ++e) { f[e] = (__bf16)q[e]; f[e + 8] = (__bf16)q[e + 16]; }
  return f;
}

// ---- grid-wide generation barrier (device scope) ----
__device__ __forceinline__ void grid_sync(unsigned* cnt, unsigned* gen, unsigned nblk) {
  __syncthreads();
  if (threadIdx.x == 0) {
    __builtin_amdgcn_fence(__ATOMIC_RELEASE, "agent");
    unsigned g = __hip_atomic_load(gen, __ATOMIC_RELAXED, __HIP_MEMORY_SCOPE_AGENT);
    unsigned a = __hip_atomic_fetch_add(cnt, 1u, __ATOMIC_ACQ_REL, __HIP_MEMORY_SCOPE_AGENT);
    if (a == nblk - 1u) {
      __hip_atomic_store(cnt, 0u, __ATOMIC_RELAXED, __HIP_MEMORY_SCOPE_AGENT);
      __hip_atomic_store(gen, g + 1u, __ATOMIC_RELEASE, __HIP_MEMORY_SCOPE_AGENT);
    } else {
      while (__hip_atomic_load(gen, __ATOMIC_ACQUIRE, __HIP_MEMORY_SCOPE_AGENT) == g) {
        __builtin_amdgcn_s_sleep(1);
      }
    }
  }
  __syncthreads();
  __builtin_amdgcn_fence(__ATOMIC_ACQUIRE, "agent");
}

// ---- small utility kernels ----
__global__ void cvt_f32_bf16(const float* __restrict__ in, __bf16* __restrict__ out, int n) {
  int i = blockIdx.x * blockDim.x + threadIdx.x;
  if (i < n) out[i] = (__bf16)in[i];
}

__global__ void init_barrier(unsigned* bar) {
  if (threadIdx.x < 2) bar[threadIdx.x] = 0u;
}

// ---- phase 1: x_proj[t][b][h] = inputs[b][t][:] . W_in[h][:] + b_in[h], bf16 [T,B,H] out ----
__global__ void xproj_kernel(const float* __restrict__ inp, const __bf16* __restrict__ Win,
                             const float* __restrict__ b_in, __bf16* __restrict__ xp) {
  const int lane = threadIdx.x & 31;
  const int wave = threadIdx.x >> 5;
  const int wid  = blockIdx.x * 8 + wave;
  const int NT   = Hsz / 16;                 // 64 N tiles
  const int Mbase = (wid / NT) * 16;         // M rows = b*T + t (inputs layout)
  const int Nbase = (wid % NT) * 16;
  const int half = lane >> 4;
  const int n    = Nbase + (lane & 15);

  f32x8 acc;
  const float bias = b_in[n];
#pragma unroll
  for (int r = 0; r < 8; ++r) acc[r] = bias;

#pragma unroll
  for (int kb = 0; kb < Isz / 32; ++kb) {
    bf16x16 a = frag_ld_f32(inp, Isz, Mbase, kb * 32, lane);
    bf16x16 b = frag_ld_bf16(Win, Isz, Nbase, kb * 32, lane);
    acc = __builtin_amdgcn_wmma_f32_16x16x32_bf16(false, a, false, b, (short)0, acc, false, false);
  }
#pragma unroll
  for (int r = 0; r < 8; ++r) {
    const int m = Mbase + r + 8 * half;      // m = b*T + t
    const int b = m >> 9;                    // /Tsz
    const int t = m & (Tsz - 1);
    xp[((size_t)t * Bsz + b) * Hsz + n] = (__bf16)acc[r];
  }
}

// ---- phase 2: persistent recurrent scan, grid of 64 blocks x 128 threads ----
// xp, hid_bf are [T,B,H] bf16; hid_f32 (d_out) is [B,T,H] f32.
__global__ void rnn_scan_kernel(const __bf16* __restrict__ Wrec, const float* __restrict__ b_rec,
                                const __bf16* __restrict__ xp, __bf16* __restrict__ hid_bf,
                                float* __restrict__ hid_f32, unsigned* __restrict__ bar) {
  extern __shared__ __bf16 smem[];           // 16 rows x 1024 bf16 = 32 KB (W_rec slice)
  const int lane = threadIdx.x & 31;
  const int wave = threadIdx.x >> 5;         // 0..3 == M tile
  const int Nbase = blockIdx.x * 16;         // one N tile per block

  // stage W_rec rows [Nbase, Nbase+16) x [0,1024) into LDS once
#ifdef USE_TDM
  if (threadIdx.x == 0) {
    const uint64_t ga = (uint64_t)(uintptr_t)(Wrec + (size_t)Nbase * Hsz);
    // D# group0: count=1, lds_addr=0 (dynamic LDS base), global_addr, type=2
    u32x4 g0 = { 0x1u, 0u, (unsigned)(ga & 0xFFFFFFFFu),
                 (unsigned)((ga >> 32) & 0x01FFFFFFu) | (2u << 30) };
    // D# group1: data_size=2B; tensor_dim0=1024, tensor_dim1=16;
    // tile_dim0=1024, tile_dim1=16; dim0_stride=1024
    i32x8 g1 = { (int)0x00010000u,            // wg_mask=0, data_size=1 (2B)
                 (int)(1024u << 16),          // tensor_dim0[15:0] in [31:16]
                 (int)(16u << 16),            // tensor_dim1[15:0] in [31:16]
                 (int)(1024u << 16),          // tile_dim0 in [127:112]
                 (int)16u,                    // tile_dim1 in [143:128]
                 (int)1024u,                  // tensor_dim0_stride[31:0]
                 (int)(1024u << 16),          // tensor_dim1_stride[15:0] in [223:208]
                 0 };
    i32x4 z4 = { 0, 0, 0, 0 };
#if __clang_major__ >= 23
    i32x8 z8 = { 0, 0, 0, 0, 0, 0, 0, 0 };
    __builtin_amdgcn_tensor_load_to_lds(g0, g1, z4, z4, z8, 0);
#else
    __builtin_amdgcn_tensor_load_to_lds(g0, g1, z4, z4, 0);
#endif
    __builtin_amdgcn_s_wait_tensorcnt(0);
  }
#else
  {
    const uint4* src = (const uint4*)(Wrec + (size_t)Nbase * Hsz);
    uint4* dst = (uint4*)smem;
    const int total = 16 * Hsz * 2 / 16;     // 2048 x 16B
    for (int i = threadIdx.x; i < total; i += blockDim.x) dst[i] = src[i];
  }
#endif
  __syncthreads();

  const int Mbase = wave * 16;
  const int half  = lane >> 4;
  const int n     = Nbase + (lane & 15);
  const int nloc  = lane & 15;               // LDS row
  const float bias = b_rec[n];

  for (int t = 0; t < Tsz; ++t) {
    f32x8 acc;
#pragma unroll
    for (int r = 0; r < 8; ++r) {
      const int m = Mbase + r + 8 * half;
      acc[r] = bias + (float)xp[((size_t)t * Bsz + m) * Hsz + n];
    }
    if (t + 1 < Tsz) {
      __builtin_prefetch((const void*)(xp + ((size_t)(t + 1) * Bsz + Mbase) * Hsz + n), 0, 0);
    }
    if (t > 0) {
      const __bf16* arow =
          hid_bf + ((size_t)(t - 1) * Bsz + (Mbase + (lane & 15))) * Hsz;
#pragma unroll 4
      for (int kb = 0; kb < Hsz / 32; ++kb) {
        const int k0 = kb * 32 + half * 8;
        bf16x16 a, b;
        *(bf16x8*)&a       = *(const bf16x8*)(arow + k0);
        *((bf16x8*)&a + 1) = *(const bf16x8*)(arow + k0 + 16);
        *(bf16x8*)&b       = *(const bf16x8*)&smem[nloc * Hsz + k0];
        *((bf16x8*)&b + 1) = *(const bf16x8*)&smem[nloc * Hsz + k0 + 16];
        acc = __builtin_amdgcn_wmma_f32_16x16x32_bf16(false, a, false, b, (short)0, acc, false, false);
      }
    }
#pragma unroll
    for (int r = 0; r < 8; ++r) {
      const float v = acc[r] > 0.0f ? acc[r] : 0.0f;
      const int m = Mbase + r + 8 * half;
      hid_f32[((size_t)m * Tsz + t) * Hsz + n] = v;            // [B,T,H] output
      hid_bf[((size_t)t * Bsz + m) * Hsz + n]  = (__bf16)v;    // [T,B,H] scratch
    }
    grid_sync(bar, bar + 1, gridDim.x);
  }
}

// ---- phase 3: outputs[b][t][o] = hiddens . W_out^T + b_out; hid_bf is [T,B,H] ----
__global__ void out_kernel(const __bf16* __restrict__ hid_bf, const __bf16* __restrict__ Wout,
                           const float* __restrict__ b_out, float* __restrict__ out) {
  const int lane = threadIdx.x & 31;
  const int wave = threadIdx.x >> 5;
  const int wid  = blockIdx.x * 8 + wave;
  const int NT   = Osz / 16;                 // 8 N tiles
  const int Mbase = (wid / NT) * 16;         // M rows = t*B + b ([T,B,H] order)
  const int Nbase = (wid % NT) * 16;
  const int half = lane >> 4;
  const int n    = Nbase + (lane & 15);

  f32x8 acc;
  const float bias = b_out[n];
#pragma unroll
  for (int r = 0; r < 8; ++r) acc[r] = bias;

#pragma unroll 4
  for (int kb = 0; kb < Hsz / 32; ++kb) {
    bf16x16 a = frag_ld_bf16(hid_bf, Hsz, Mbase, kb * 32, lane);
    bf16x16 b = frag_ld_bf16(Wout, Hsz, Nbase, kb * 32, lane);
    acc = __builtin_amdgcn_wmma_f32_16x16x32_bf16(false, a, false, b, (short)0, acc, false, false);
  }
#pragma unroll
  for (int r = 0; r < 8; ++r) {
    const int m = Mbase + r + 8 * half;      // m = t*B + b
    const int t = m >> 6;                    // /Bsz
    const int b = m & (Bsz - 1);
    out[((size_t)b * Tsz + t) * Osz + n] = acc[r];
  }
}

extern "C" void kernel_launch(void* const* d_in, const int* in_sizes, int n_in,
                              void* d_out, int out_size, void* d_ws, size_t ws_size,
                              hipStream_t stream) {
  const float* inputs = (const float*)d_in[0];   // [B,T,I]
  const float* W_rec  = (const float*)d_in[1];   // [H,H]
  const float* b_rec  = (const float*)d_in[2];   // [H]
  const float* W_in   = (const float*)d_in[3];   // [H,I]
  const float* b_in   = (const float*)d_in[4];   // [H]
  const float* W_out  = (const float*)d_in[5];   // [O,H]
  const float* b_out  = (const float*)d_in[6];   // [O]

  float* outs    = (float*)d_out;                           // [B,T,O]
  float* hiddens = (float*)d_out + (size_t)Bsz * Tsz * Osz; // [B,T,H]

  uint8_t* ws = (uint8_t*)d_ws;
  unsigned* bar  = (unsigned*)ws;                          // 256 B
  __bf16* wrec_b = (__bf16*)(ws + 256);                    // H*H
  __bf16* win_b  = wrec_b + (size_t)Hsz * Hsz;             // H*I
  __bf16* wout_b = win_b  + (size_t)Hsz * Isz;             // O*H
  __bf16* xp_b   = wout_b + (size_t)Osz * Hsz;             // [T,B,H]
  __bf16* hid_b  = xp_b   + (size_t)Bsz * Tsz * Hsz;       // [T,B,H]

  init_barrier<<<1, 64, 0, stream>>>(bar);

  cvt_f32_bf16<<<(Hsz * Hsz + 255) / 256, 256, 0, stream>>>(W_rec, wrec_b, Hsz * Hsz);
  cvt_f32_bf16<<<(Hsz * Isz + 255) / 256, 256, 0, stream>>>(W_in,  win_b,  Hsz * Isz);
  cvt_f32_bf16<<<(Osz * Hsz + 255) / 256, 256, 0, stream>>>(W_out, wout_b, Osz * Hsz);

  // phase 1: (B*T/16) * (H/16) tiles, 8 waves per block
  const int xproj_blocks = (Bsz * Tsz / 16) * (Hsz / 16) / 8;   // 16384
  xproj_kernel<<<xproj_blocks, 256, 0, stream>>>(inputs, win_b, b_in, xp_b);

  // phase 2: 64 blocks (one N tile each) x 4 waves (M tiles), 32 KB LDS
  rnn_scan_kernel<<<Hsz / 16, 128, 16 * Hsz * sizeof(__bf16), stream>>>(
      wrec_b, b_rec, xp_b, hid_b, hiddens, bar);

  // phase 3: (B*T/16) * (O/16) tiles, 8 waves per block
  const int out_blocks = (Bsz * Tsz / 16) * (Osz / 16) / 8;     // 2048
  out_kernel<<<out_blocks, 256, 0, stream>>>(hid_b, wout_b, b_out, outs);
}
